// TabularGNN_46462956208473
// MI455X (gfx1250) — compile-verified, matrix operand order checked by reading, not run on hardware
//
#include <hip/hip_runtime.h>
#include <hip/hip_bf16.h>

// ---------------------------------------------------------------------------
// GCN (4x GCNConv + log_softmax) for MI455X / gfx1250.
//   Dense path:  bf16 WMMA 16x16x32 (fp32 accum). B-strip staged in LDS once
//                per block via GLOBAL_LOAD_ASYNC_TO_LDS_B128 (ASYNCcnt), A
//                double-buffered in registers. All tiles L2-resident.
//   Sparse path: edge-parallel gather * norm -> f32 atomics into AGG.
//   Epilogues fuse self-loop term, bias, ReLU, and bf16 repack.
// ---------------------------------------------------------------------------

#define HID 246      // hidden width
#define LD  256      // padded leading dimension (cols) for all node matrices
#define KPAD 256     // padded K for every GEMM
#define BROW 264     // LDS row stride in bf16 elems (528 B: 132 dw = 4 mod 64 banks)

typedef __attribute__((ext_vector_type(16))) __bf16 v16bf;
typedef __attribute__((ext_vector_type(8)))  __bf16 v8bf;
typedef __attribute__((ext_vector_type(8)))  float  v8f;
typedef __attribute__((ext_vector_type(4)))  int    v4i;
typedef __attribute__((ext_vector_type(4)))  unsigned int v4u;

#if defined(__has_builtin)
#if __has_builtin(__builtin_amdgcn_global_load_async_to_lds_b128)
#define USE_ASYNC_LDS 1
#endif
#endif

static __device__ __forceinline__ unsigned short f2bf(float f) {
    // round-to-nearest-even fp32 -> bf16
    unsigned int u = __float_as_uint(f);
    unsigned int r = (u + 0x7FFFu + ((u >> 16) & 1u)) >> 16;
    return (unsigned short)r;
}

// ---- degree / norm --------------------------------------------------------
__global__ void deg_kernel(const int* __restrict__ ei, int E,
                           float* __restrict__ deg, int N) {
    int i = blockIdx.x * blockDim.x + threadIdx.x;
    if (i < E) atomicAdd(&deg[ei[E + i]], 1.0f);   // dst column of edge_index
    if (i < N) atomicAdd(&deg[i], 1.0f);           // self loop
}

__global__ void dinv_kernel(float* __restrict__ deg, int N) {
    int i = blockIdx.x * blockDim.x + threadIdx.x;
    if (i < N) {
        float d = deg[i];
        deg[i] = (d > 0.0f) ? rsqrtf(fmaxf(d, 1.0f)) : 0.0f;   // in place -> dinv
    }
}

// ---- conversions ----------------------------------------------------------
// W is [Fin, HID] row-major fp32 -> Wt is [LD(=Nout), KPAD] row-major bf16 (transposed, zero-padded)
__global__ void convert_w_kernel(const float* __restrict__ W, int Fin,
                                 unsigned short* __restrict__ Wt) {
    int n = blockIdx.x;      // 0..255 : output feature (row of Wt)
    int k = threadIdx.x;     // 0..255 : input feature
    float v = (n < HID && k < Fin) ? W[k * HID + n] : 0.0f;
    Wt[n * KPAD + k] = f2bf(v);
}

__global__ void convert_x_kernel(const float* __restrict__ x,
                                 unsigned short* __restrict__ A, long long n) {
    long long i = (long long)blockIdx.x * blockDim.x + threadIdx.x;
    if (i < n) A[i] = f2bf(x[i]);
}

// ---- WMMA GEMM:  H[M,LD] = A[M,KPAD](bf16) * Wt^T  (Wt stored N-major) -----
// Block = 8 waves; block tile = 128(M) x 64(N); B strip staged in LDS.
__global__ void __launch_bounds__(256)
gemm_bf16_wmma(const unsigned short* __restrict__ A,
               const unsigned short* __restrict__ Bt,   // [LD, KPAD], row n = col n of W
               float* __restrict__ C, int M) {
    __shared__ __align__(16) unsigned short sB[64 * BROW];   // 33 KB

    const int tid = threadIdx.x;
    const int n0  = blockIdx.y * 64;

    // ---- stage B strip [64 x 256] bf16 -> LDS (2048 x 16B chunks, 8/thread)
#pragma unroll
    for (int c = tid; c < 2048; c += 256) {
        const int row = c >> 5;        // 0..63
        const int seg = c & 31;        // 16B segment within row
        const unsigned short* g = Bt + (size_t)(n0 + row) * KPAD + seg * 8;
        unsigned short*       l = &sB[row * BROW + seg * 8];
#ifdef USE_ASYNC_LDS
        __builtin_amdgcn_global_load_async_to_lds_b128(
            (__attribute__((address_space(1))) v4i*)g,
            (__attribute__((address_space(3))) v4i*)l, 0, 0);
#else
        *(v4u*)l = *(const v4u*)g;
#endif
    }
#ifdef USE_ASYNC_LDS
#if __has_builtin(__builtin_amdgcn_s_wait_asynccnt)
    __builtin_amdgcn_s_wait_asynccnt(0);
#else
    asm volatile("s_wait_asynccnt 0x0" ::: "memory");
#endif
#endif
    __syncthreads();

    const int wave  = tid >> 5;
    const int lane  = tid & 31;
    const int mtile = blockIdx.x * 8 + wave;
    if (mtile * 16 >= M) return;                 // after barrier; wave-uniform
    const int m0   = mtile * 16;
    const int half = lane >> 4;                  // 0: K 0..7/16..23, 1: K 8..15/24..31
    const int lrow = lane & 15;

    const __bf16* Arow = (const __bf16*)(A + (size_t)(m0 + lrow) * KPAD);
    const unsigned short* Br0 = &sB[( 0 + lrow) * BROW];
    const unsigned short* Br1 = &sB[(16 + lrow) * BROW];
    const unsigned short* Br2 = &sB[(32 + lrow) * BROW];
    const unsigned short* Br3 = &sB[(48 + lrow) * BROW];

    v8f acc0 = {}, acc1 = {}, acc2 = {}, acc3 = {};

    // A double-buffer: prefetch next k-step while WMMAs consume current
    v8bf alo = *(const v8bf*)(Arow + half * 8);
    v8bf ahi = *(const v8bf*)(Arow + 16 + half * 8);

#pragma unroll
    for (int k0 = 0; k0 < KPAD; k0 += 32) {
        v8bf nlo = alo, nhi = ahi;
        if (k0 + 32 < KPAD) {
            nlo = *(const v8bf*)(Arow + (k0 + 32) + half * 8);
            nhi = *(const v8bf*)(Arow + (k0 + 48) + half * 8);
        }
        const int klo = k0 + half * 8;
        const int khi = k0 + 16 + half * 8;

        v16bf a = __builtin_shufflevector(alo, ahi, 0,1,2,3,4,5,6,7,8,9,10,11,12,13,14,15);

        // conflict-free ds_load_b128 pairs (row stride 132 dw => 4-bank lane skew)
        v8bf b0l = *(const v8bf*)((const __bf16*)Br0 + klo), b0h = *(const v8bf*)((const __bf16*)Br0 + khi);
        v8bf b1l = *(const v8bf*)((const __bf16*)Br1 + klo), b1h = *(const v8bf*)((const __bf16*)Br1 + khi);
        v8bf b2l = *(const v8bf*)((const __bf16*)Br2 + klo), b2h = *(const v8bf*)((const __bf16*)Br2 + khi);
        v8bf b3l = *(const v8bf*)((const __bf16*)Br3 + klo), b3h = *(const v8bf*)((const __bf16*)Br3 + khi);
        v16bf b0 = __builtin_shufflevector(b0l, b0h, 0,1,2,3,4,5,6,7,8,9,10,11,12,13,14,15);
        v16bf b1 = __builtin_shufflevector(b1l, b1h, 0,1,2,3,4,5,6,7,8,9,10,11,12,13,14,15);
        v16bf b2 = __builtin_shufflevector(b2l, b2h, 0,1,2,3,4,5,6,7,8,9,10,11,12,13,14,15);
        v16bf b3 = __builtin_shufflevector(b3l, b3h, 0,1,2,3,4,5,6,7,8,9,10,11,12,13,14,15);

        acc0 = __builtin_amdgcn_wmma_f32_16x16x32_bf16(false, a, false, b0, (short)0, acc0, false, false);
        acc1 = __builtin_amdgcn_wmma_f32_16x16x32_bf16(false, a, false, b1, (short)0, acc1, false, false);
        acc2 = __builtin_amdgcn_wmma_f32_16x16x32_bf16(false, a, false, b2, (short)0, acc2, false, false);
        acc3 = __builtin_amdgcn_wmma_f32_16x16x32_bf16(false, a, false, b3, (short)0, acc3, false, false);

        alo = nlo; ahi = nhi;
    }

    // f32 C/D layout: VGPR r -> row m0 + r + 8*half, col n0 + lrow (+16 per tile)
#pragma unroll
    for (int r = 0; r < 8; ++r) {
        const size_t m = (size_t)(m0 + r + half * 8);
        C[m * LD + n0 +  0 + lrow] = acc0[r];
        C[m * LD + n0 + 16 + lrow] = acc1[r];
        C[m * LD + n0 + 32 + lrow] = acc2[r];
        C[m * LD + n0 + 48 + lrow] = acc3[r];
    }
}

// ---- sparse neighbor aggregation: AGG[dst] += H[src] * dinv[src]*dinv[dst] -
__global__ void agg_kernel(const int* __restrict__ ei, int E,
                           const float* __restrict__ H,
                           const float* __restrict__ dinv,
                           float* __restrict__ AGG) {
    const int e = blockIdx.x;
    const int f = threadIdx.x;
    if (f >= HID) return;
    const int s = ei[e];            // scalar (block-uniform) loads
    const int d = ei[E + e];
    const float nrm = dinv[s] * dinv[d];
    atomicAdd(&AGG[(size_t)d * LD + f], H[(size_t)s * LD + f] * nrm);
}

// ---- epilogues ------------------------------------------------------------
// layers 1..3: out = relu(AGG + H*dinv^2 + b) -> bf16 A for next GEMM (zero-padded)
__global__ void epilogue_relu_bf16(const float* __restrict__ H,
                                   const float* __restrict__ AGG,
                                   const float* __restrict__ dinv,
                                   const float* __restrict__ bias,
                                   unsigned short* __restrict__ A) {
    const int node = blockIdx.x;
    const int f = threadIdx.x;
    const float di = dinv[node];
    float v = 0.0f;
    if (f < HID)
        v = AGG[(size_t)node * LD + f] + H[(size_t)node * LD + f] * di * di + bias[f];
    v = fmaxf(v, 0.0f);
    A[(size_t)node * LD + f] = f2bf(v);   // cols 246..255 become 0
}

// layer 4: AGG <- AGG + H*dinv^2 + b  (fp32, in place, no relu)
__global__ void epilogue_final(const float* __restrict__ H,
                               float* __restrict__ AGG,
                               const float* __restrict__ dinv,
                               const float* __restrict__ bias) {
    const int node = blockIdx.x;
    const int f = threadIdx.x;
    if (f >= HID) return;
    const float di = dinv[node];
    AGG[(size_t)node * LD + f] =
        AGG[(size_t)node * LD + f] + H[(size_t)node * LD + f] * di * di + bias[f];
}

// ---- log_softmax over HID features ----------------------------------------
__global__ void logsoftmax_kernel(const float* __restrict__ X,
                                  float* __restrict__ out) {
    __shared__ float red[256];
    const int node = blockIdx.x;
    const int f = threadIdx.x;
    const float v = (f < HID) ? X[(size_t)node * LD + f] : -3.4e38f;
    red[f] = v;
    __syncthreads();
#pragma unroll
    for (int s = 128; s > 0; s >>= 1) {
        if (f < s) red[f] = fmaxf(red[f], red[f + s]);
        __syncthreads();
    }
    const float m = red[0];
    __syncthreads();
    red[f] = (f < HID) ? __expf(v - m) : 0.0f;
    __syncthreads();
#pragma unroll
    for (int s = 128; s > 0; s >>= 1) {
        if (f < s) red[f] += red[f + s];
        __syncthreads();
    }
    const float lse = __logf(red[0]);
    if (f < HID) out[(size_t)node * HID + f] = v - m - lse;
}

// ---------------------------------------------------------------------------
extern "C" void kernel_launch(void* const* d_in, const int* in_sizes, int n_in,
                              void* d_out, int out_size, void* d_ws, size_t ws_size,
                              hipStream_t stream) {
    const float* x  = (const float*)d_in[0];
    const int*   ei = (const int*)d_in[1];
    const float* W[4] = { (const float*)d_in[2], (const float*)d_in[4],
                          (const float*)d_in[6], (const float*)d_in[8] };
    const float* B[4] = { (const float*)d_in[3], (const float*)d_in[5],
                          (const float*)d_in[7], (const float*)d_in[9] };
    const int N = in_sizes[0] / 256;     // 50000
    const int E = in_sizes[1] / 2;       // 320000
    float* out = (float*)d_out;

    // workspace carve-out (~129 MB)
    char*  ws  = (char*)d_ws;
    size_t off = 0;
    auto carve = [&](size_t bytes) -> void* {
        void* p = ws + off;
        off = (off + bytes + 255) & ~(size_t)255;
        return p;
    };
    float*          dinv = (float*)         carve((size_t)N * sizeof(float));
    unsigned short* A    = (unsigned short*)carve((size_t)N * LD * sizeof(unsigned short));
    float*          H    = (float*)         carve((size_t)N * LD * sizeof(float));
    float*          AGG  = (float*)         carve((size_t)N * LD * sizeof(float));
    unsigned short* Wt   = (unsigned short*)carve((size_t)4 * LD * KPAD * sizeof(unsigned short));
    (void)ws_size; (void)n_in; (void)out_size;

    // 1) degree -> dinv
    (void)hipMemsetAsync(dinv, 0, (size_t)N * sizeof(float), stream);
    {
        int work = (E > N) ? E : N;
        deg_kernel<<<(work + 255) / 256, 256, 0, stream>>>(ei, E, dinv, N);
        dinv_kernel<<<(N + 255) / 256, 256, 0, stream>>>(dinv, N);
    }

    // 2) weights -> transposed, padded bf16
    convert_w_kernel<<<LD, 256, 0, stream>>>(W[0], 256, Wt + 0 * LD * KPAD);
    convert_w_kernel<<<LD, 256, 0, stream>>>(W[1], HID, Wt + 1 * LD * KPAD);
    convert_w_kernel<<<LD, 256, 0, stream>>>(W[2], HID, Wt + 2 * LD * KPAD);
    convert_w_kernel<<<LD, 256, 0, stream>>>(W[3], HID, Wt + 3 * LD * KPAD);

    // 3) x -> bf16 A (layer-1 operand; K=256 exactly, no padding needed)
    {
        long long n = (long long)N * 256;
        convert_x_kernel<<<(unsigned)((n + 255) / 256), 256, 0, stream>>>(x, A, n);
    }

    // 4) layers
    const int mtiles = (N + 15) / 16;                    // 3125
    dim3 ggrid((mtiles + 7) / 8, LD / 64);               // (391, 4)
    for (int l = 0; l < 4; ++l) {
        gemm_bf16_wmma<<<ggrid, 256, 0, stream>>>(A, Wt + (size_t)l * LD * KPAD, H, N);
        (void)hipMemsetAsync(AGG, 0, (size_t)N * LD * sizeof(float), stream);
        agg_kernel<<<E, 256, 0, stream>>>(ei, E, H, dinv, AGG);
        if (l < 3)
            epilogue_relu_bf16<<<N, 256, 0, stream>>>(H, AGG, dinv, B[l], A);
        else
            epilogue_final<<<N, 256, 0, stream>>>(H, AGG, dinv, B[l]);
    }

    // 5) log_softmax -> d_out [N, 246]
    logsoftmax_kernel<<<N, 256, 0, stream>>>(AGG, out);
}
